// VGGExtractorNP_11166914970532
// MI455X (gfx1250) — matrix operand order repaired and checked by reading, not compiled, not run on hardware
//
#include <hip/hip_runtime.h>
#include <hip/hip_bf16.h>

// ---------------------------------------------------------------------------
// VGG keypoint extractor (SuperPoint-like) for MI455X / gfx1250.
// Conv layers: implicit GEMM with v_wmma_f32_16x16x32_f16 (wave32),
// weight panels staged LDS via the Tensor Data Mover (tensor_load_to_lds).
// ---------------------------------------------------------------------------

typedef __attribute__((ext_vector_type(16))) _Float16 v16h;
typedef __attribute__((ext_vector_type(8)))  _Float16 v8h;
typedef __attribute__((ext_vector_type(8)))  float    v8f;
typedef unsigned int u32x4 __attribute__((ext_vector_type(4)));
typedef int          i32x4 __attribute__((ext_vector_type(4)));
typedef int          i32x8 __attribute__((ext_vector_type(8)));

#if defined(__HIP_DEVICE_COMPILE__) && __has_builtin(__builtin_amdgcn_tensor_load_to_lds) && \
    __has_builtin(__builtin_amdgcn_s_wait_tensorcnt)
#define HAVE_TDM 1
#else
#define HAVE_TDM 0
#endif

constexpr int  Bn    = 2;
constexpr int  H     = 320;
constexpr int  W     = 416;
constexpr int  HW    = H * W;          // 133120
constexpr int  TOPK  = 2048;
constexpr int  BORD  = 4;
constexpr int  NMSR  = 4;
constexpr float THRESH = 5e-4f;
constexpr float BNEPS  = 1e-5f;

// output section offsets (floats), in reference tuple return order
constexpr size_t O_SCORE  = 0;                                // [B,1,H,W]
constexpr size_t O_NMS    = O_SCORE  + (size_t)Bn*HW;         // [B,H,W]
constexpr size_t O_NDESC  = O_NMS    + (size_t)Bn*HW;         // [B,128,H,W]
constexpr size_t O_DDESC  = O_NDESC  + (size_t)Bn*128*HW;     // [B,HW,128]
constexpr size_t O_SPOS   = O_DDESC  + (size_t)Bn*HW*128;     // [B,K,3]
constexpr size_t O_SDESC  = O_SPOS   + (size_t)Bn*TOPK*3;     // [B,K,128]
constexpr size_t O_DPOS   = O_SDESC  + (size_t)Bn*TOPK*128;   // [B,HW,3]

// ---------------------------------------------------------------------------
// Weight prep: OIHW f32 -> B-matrix f16 [COUTP][KPAD] (k = tap*CIN + ci),
// fold conv-bias + BN into per-channel scale/bias.
// ---------------------------------------------------------------------------
__global__ void k_prep_weights(const float* __restrict__ Wf,
                               const float* __restrict__ bb,
                               const float* __restrict__ g,
                               const float* __restrict__ bt,
                               const float* __restrict__ mu,
                               const float* __restrict__ var,
                               _Float16* __restrict__ wh,
                               float* __restrict__ scale,
                               float* __restrict__ bias,
                               int CIN, int COUT, int KSZ, int COUTP, int KPAD)
{
    int i = blockIdx.x * blockDim.x + threadIdx.x;
    int total = COUTP * KPAD;
    if (i < total) {
        int o = i / KPAD, k = i - o * KPAD;
        int KTOT = KSZ * KSZ * CIN;
        float v = 0.f;
        if (o < COUT && k < KTOT) {
            int tap = (KSZ == 1) ? 0 : (k / CIN);
            int ci  = (KSZ == 1) ? k : (k - tap * CIN);
            int ky  = tap / KSZ, kx = tap % KSZ;
            v = Wf[(((size_t)o * CIN + ci) * KSZ + ky) * KSZ + kx];
        }
        wh[i] = (_Float16)v;
    }
    if (i < COUTP) {
        float s = 0.f, bo = 0.f;
        if (i < COUT) {
            float inv = g[i] * rsqrtf(var[i] + BNEPS);
            s  = inv;
            bo = bb[i] * inv + bt[i] - mu[i] * inv;
        }
        scale[i] = s;
        bias[i]  = bo;
    }
}

__global__ void k_x_to_h(const float* __restrict__ x, _Float16* __restrict__ xh)
{
    int i = blockIdx.x * blockDim.x + threadIdx.x;
    if (i < Bn * HW) xh[i] = (_Float16)x[i];
}

// ---------------------------------------------------------------------------
// Implicit-GEMM conv + BN (+ReLU) using WMMA f32_16x16x32_f16.
// Block: 256 threads = 8 waves; block tile = 128 pixels; each wave owns a
// 16-pixel M subtile; N loop over 16-channel tiles; K in 32-wide chunks.
// B panel (16 outch x KPAD) is staged into LDS once per N-tile by the TDM.
// Activations NHWC f16: act[((b*HW + y*W + x)*CIN) + ci].
// ---------------------------------------------------------------------------
template<int CIN, int COUT, int KSZ, bool RELU, bool OUT_F32>
__global__ __launch_bounds__(256) void k_conv_wmma(
    const _Float16* __restrict__ act_in,
    const _Float16* __restrict__ wh,      // [COUTP][KPAD] f16
    const float*    __restrict__ scale,   // [COUTP]
    const float*    __restrict__ bias,    // [COUTP]
    _Float16*       __restrict__ act_out, // NHWC f16 (if !OUT_F32)
    float*          __restrict__ out_f32) // pixel-major f32 (if OUT_F32)
{
    constexpr int KTOT = KSZ * KSZ * CIN;
    constexpr int KPAD = ((KTOT + 31) / 32) * 32;
    constexpr int NCH  = KPAD / 32;
    constexpr int NT   = (COUT + 15) / 16;

    __shared__ __attribute__((aligned(16))) _Float16 As[128 * 32];  // 8 KB
    __shared__ __attribute__((aligned(16))) _Float16 Bs[16 * KPAD]; // <= 36 KB

    const int b    = blockIdx.z;
    const int p0   = blockIdx.x * 128;
    const int tid  = threadIdx.x;
    const int wave = tid >> 5;
    const int lane = tid & 31;
    const int m    = lane & 15;   // A row / B column / D column index
    const int hl   = lane >> 4;   // lane half selector

    if (lane == 0)
        __builtin_prefetch(act_in + ((size_t)b * HW + p0) * CIN, 0, 1);

    for (int nt = 0; nt < NT; ++nt) {
        // ---- stage B panel: 16 rows x KPAD halves, via Tensor Data Mover --
#if HAVE_TDM
        if (tid < 32) {   // wave 0 issues the DMA (EXEC ignored by TDM)
            unsigned long long ga =
                (unsigned long long)(const void*)(wh + (size_t)(nt * 16) * KPAD);
            unsigned lds_off = (unsigned)(unsigned long long)(const void*)&Bs[0];
            u32x4 g0;
            g0[0] = 1u;                                   // count=1, user desc
            g0[1] = lds_off;                              // lds_addr
            g0[2] = (unsigned)ga;                         // global_addr[31:0]
            g0[3] = ((unsigned)(ga >> 32) & 0x01FFFFFFu)  // global_addr[56:32]
                    | (2u << 30);                         // type=2 ("image")
            i32x8 g1;
            g1[0] = 1 << 16;                              // data_size=2B
            g1[1] = (int)(((unsigned)KPAD & 0xFFFFu) << 16); // tensor_dim0 lo16
            g1[2] = (int)(((unsigned)KPAD >> 16) | (16u << 16)); // td0 hi | td1 lo
            g1[3] = (int)((unsigned)KPAD << 16);          // tile_dim0 = KPAD
            g1[4] = 16;                                   // tile_dim1 = 16 rows
            g1[5] = KPAD;                                 // tensor_dim0_stride lo
            g1[6] = 0;
            g1[7] = 0;
            i32x4 gz = {0, 0, 0, 0};
#if defined(__clang_major__) && (__clang_major__ >= 23)
            i32x8 gz8 = {0, 0, 0, 0, 0, 0, 0, 0};
            __builtin_amdgcn_tensor_load_to_lds(g0, g1, gz, gz, gz8, 0);
#else
            __builtin_amdgcn_tensor_load_to_lds(g0, g1, gz, gz, 0);
#endif
        }
        __builtin_amdgcn_s_wait_tensorcnt(0);
#else
        // fallback: cooperative vector copy
        for (int e = tid * 8; e < 16 * KPAD; e += 256 * 8) {
            int n = e / KPAD, off = e - n * KPAD;
            *(v8h*)&Bs[e] = *(const v8h*)(wh + (size_t)(nt * 16 + n) * KPAD + off);
        }
#endif
        __syncthreads();

        v8f acc = {};
        for (int kc = 0; kc < NCH; ++kc) {
            // ---- stage A chunk: 128 pixels x 32 k (implicit im2col + pad) --
            if (CIN >= 32) {
                // 16 consecutive k == 16 consecutive channels of one tap ->
                // two b128 global loads + two b128 LDS stores per thread
                int e   = tid * 16;
                int pix = e >> 5;
                int kk0 = e & 31;
                int k0  = kc * 32 + kk0;
                int tap = (KSZ == 1) ? 0 : (k0 / CIN);
                int ci  = (KSZ == 1) ? k0 : (k0 - tap * CIN);
                int p   = p0 + pix;
                int y   = p / W + (tap / KSZ) - (KSZ / 2);
                int x   = p % W + (tap % KSZ) - (KSZ / 2);
                v8h d0 = {}, d1 = {};
                if (y >= 0 && y < H && x >= 0 && x < W) {
                    const _Float16* src =
                        act_in + ((size_t)b * HW + (size_t)y * W + x) * CIN + ci;
                    d0 = *(const v8h*)src;
                    d1 = *(const v8h*)(src + 8);
                }
                *(v8h*)&As[e]     = d0;
                *(v8h*)&As[e + 8] = d1;
            } else {
                #pragma unroll
                for (int i = 0; i < 16; ++i) {
                    int e   = tid * 16 + i;
                    int pix = e >> 5;
                    int kk  = e & 31;
                    int k   = kc * 32 + kk;
                    _Float16 v = (_Float16)0.f;
                    if (k < KTOT) {
                        int tap = (KSZ == 1) ? 0 : (k / CIN);
                        int ci  = (KSZ == 1) ? k : (k - tap * CIN);
                        int p = p0 + pix;
                        int y = p / W + (tap / KSZ) - (KSZ / 2);
                        int x = p % W + (tap % KSZ) - (KSZ / 2);
                        if (y >= 0 && y < H && x >= 0 && x < W)
                            v = act_in[((size_t)b * HW + (size_t)y * W + x) * CIN + ci];
                    }
                    As[e] = v;
                }
            }
            __syncthreads();

            // ---- A fragment (16x32 f16): lanes 0-15 K{0..15},
            //      lanes 16-31 K{16..31}, row m per lane ----
            const _Float16* arow = &As[(wave * 16 + m) * 32];
            v8h a0 = *(const v8h*)(arow + hl * 16);
            v8h a1 = *(const v8h*)(arow + hl * 16 + 8);
            v16h afrag = __builtin_shufflevector(a0, a1,
                0,1,2,3,4,5,6,7,8,9,10,11,12,13,14,15);

            // ---- B fragment (32x16 f16): lane half selects K 0..15/16..31 --
            const _Float16* brow = &Bs[m * KPAD + kc * 32];
            v8h b0 = *(const v8h*)(brow + hl * 16);
            v8h b1 = *(const v8h*)(brow + hl * 16 + 8);
            v16h bfrag = __builtin_shufflevector(b0, b1,
                0,1,2,3,4,5,6,7,8,9,10,11,12,13,14,15);

            acc = __builtin_amdgcn_wmma_f32_16x16x32_f16(
                false, afrag, false, bfrag, (short)0, acc, false, false);
            __syncthreads();
        }
        // ---- epilogue: fused BN scale/bias (+ReLU), scatter per D layout --
        int n_ch = nt * 16 + m;
        if (n_ch < COUT) {
            float sc = scale[n_ch], bo = bias[n_ch];
            #pragma unroll
            for (int r = 0; r < 8; ++r) {
                int mm = r + 8 * hl;             // D: VGPR r -> M=r / M=r+8
                int p  = p0 + wave * 16 + mm;
                float v = acc[r] * sc + bo;
                if (RELU) v = fmaxf(v, 0.f);
                if (OUT_F32)
                    out_f32[((size_t)b * HW + p) * COUT + n_ch] = v;
                else
                    act_out[((size_t)b * HW + p) * COUT + n_ch] = (_Float16)v;
            }
        }
    }
}

// ---------------------------------------------------------------------------
// Post-processing kernels
// ---------------------------------------------------------------------------
__global__ void k_score(const float* __restrict__ logits,
                        float* __restrict__ out_score,
                        float* __restrict__ dense_pos)
{
    int i = blockIdx.x * blockDim.x + threadIdx.x;
    if (i >= Bn * HW) return;
    float p = 1.f / (1.f + expf(-logits[i]));
    out_score[i] = p;
    int pp = i % HW;
    dense_pos[(size_t)i * 3 + 0] = (float)(pp / W);
    dense_pos[(size_t)i * 3 + 1] = (float)(pp % W);
    dense_pos[(size_t)i * 3 + 2] = p;
}

__global__ void k_nms(const float* __restrict__ score, float* __restrict__ nmsbuf)
{
    int i = blockIdx.x * blockDim.x + threadIdx.x;
    if (i >= Bn * HW) return;
    int b = i / HW, p = i % HW, y = p / W, x = p % W;
    float s = score[i], mx = s;
    for (int dy = -NMSR; dy <= NMSR; ++dy) {
        int yy = y + dy;
        if (yy < 0 || yy >= H) continue;
        for (int dx = -NMSR; dx <= NMSR; ++dx) {
            int xx = x + dx;
            if (xx < 0 || xx >= W) continue;
            mx = fmaxf(mx, score[(size_t)b * HW + yy * W + xx]);
        }
    }
    float v = (s == mx && s > THRESH) ? s : 0.f;
    if (y < BORD || y >= H - BORD || x < BORD || x >= W - BORD) v = 0.f;
    nmsbuf[i] = v;
}

__global__ void k_fill_f32(float* p, float v, int n)
{
    int i = blockIdx.x * blockDim.x + threadIdx.x;
    if (i < n) p[i] = v;
}
__global__ void k_fill_u32(unsigned* p, unsigned v, int n)
{
    int i = blockIdx.x * blockDim.x + threadIdx.x;
    if (i < n) p[i] = v;
}

// positive-float histogram on bits[30:19] (order-preserving for v>0)
__global__ void k_hist(const float* __restrict__ nmsbuf, unsigned* __restrict__ hist)
{
    int i = blockIdx.x * blockDim.x + threadIdx.x;
    if (i >= Bn * HW) return;
    float v = nmsbuf[i];
    if (v > 0.f) {
        unsigned bin = __float_as_uint(v) >> 19;   // <= 0x7F0 for v<=1
        atomicAdd(&hist[(i / HW) * 2048 + bin], 1u);
    }
}

__global__ void k_select(const unsigned* __restrict__ hist, int* __restrict__ ctrl)
{
    int b = blockIdx.x;
    const unsigned* h = hist + b * 2048;
    int* c = ctrl + b * 8;
    int cum = 0, Tb = -1;
    for (int bin = 2047; bin >= 0; --bin) {
        int cc = (int)h[bin];
        if (cum + cc >= TOPK) { Tb = bin; break; }
        cum += cc;
    }
    c[0] = Tb;          // threshold bin (-1: take everything positive)
    c[1] = cum;         // count strictly above threshold bin
    c[2] = TOPK - cum;  // budget within threshold bin
    c[3] = 0;           // above counter
    c[4] = 0;           // equal counter
}

__global__ void k_collect(const float* __restrict__ nmsbuf, int* __restrict__ ctrl,
                          int* __restrict__ idx, float* __restrict__ val,
                          float* __restrict__ out_nms)
{
    int i = blockIdx.x * blockDim.x + threadIdx.x;
    if (i >= Bn * HW) return;
    float v = nmsbuf[i];
    if (v <= 0.f) return;
    int b = i / HW, p = i % HW;
    int* c = ctrl + b * 8;
    int Tb = c[0], above = c[1], rem = c[2];
    int bin = (int)(__float_as_uint(v) >> 19);
    int slot = -1;
    if (bin > Tb) {
        slot = atomicAdd(&c[3], 1);
    } else if (bin == Tb) {
        int e = atomicAdd(&c[4], 1);
        if (e < rem) slot = above + e;
    }
    if (slot >= 0 && slot < TOPK) {
        idx[b * TOPK + slot] = p;
        val[b * TOPK + slot] = v;
        out_nms[i] = v;
    }
}

__global__ void k_fillrest(const int* __restrict__ ctrl,
                           int* __restrict__ idx, float* __restrict__ val)
{
    int s = blockIdx.x * blockDim.x + threadIdx.x;
    int b = blockIdx.y;
    if (s >= TOPK) return;
    const int* c = ctrl + b * 8;
    int eq = c[4] < c[2] ? c[4] : c[2];
    int filled = c[1] + eq;
    if (s >= filled) { idx[b * TOPK + s] = s; val[b * TOPK + s] = 0.f; }
}

__global__ void k_sparse_pos(const int* __restrict__ idx, const float* __restrict__ val,
                             float* __restrict__ out4)
{
    int s = blockIdx.x * blockDim.x + threadIdx.x;
    int b = blockIdx.y;
    if (s >= TOPK) return;
    int p = idx[b * TOPK + s];
    size_t o = ((size_t)b * TOPK + s) * 3;
    out4[o + 0] = (float)(p / W);
    out4[o + 1] = (float)(p % W);
    out4[o + 2] = val[b * TOPK + s];
}

__global__ void k_sparse_desc(const int* __restrict__ idx, const float* __restrict__ raw,
                              const float* __restrict__ scl, float* __restrict__ out5)
{
    int s = blockIdx.x, b = blockIdx.y, c = threadIdx.x;
    int p = idx[b * TOPK + s];
    float v = raw[((size_t)b * HW + p) * 128 + c];
    __shared__ float red[128];
    red[c] = v * v;
    __syncthreads();
    for (int o = 64; o > 0; o >>= 1) {
        if (c < o) red[c] += red[c + o];
        __syncthreads();
    }
    float nrm = fmaxf(sqrtf(red[0]), 1e-12f);
    out5[((size_t)b * TOPK + s) * 128 + c] = scl[0] * v / nrm;
}

__global__ void k_dense_desc(const float* __restrict__ raw, const float* __restrict__ scl,
                             float* __restrict__ out_norm,   // [B,128,HW]
                             float* __restrict__ out_dense)  // [B,HW,128]
{
    int p = blockIdx.x, b = blockIdx.y, c = threadIdx.x;
    float v = raw[((size_t)b * HW + p) * 128 + c];
    __shared__ float red[128];
    red[c] = v * v;
    __syncthreads();
    for (int o = 64; o > 0; o >>= 1) {
        if (c < o) red[c] += red[c + o];
        __syncthreads();
    }
    float nrm = fmaxf(sqrtf(red[0]), 1e-12f);
    float nv  = scl[0] * v / nrm;
    out_norm[((size_t)(b * 128 + c)) * HW + p] = nv;
    out_dense[((size_t)b * HW + p) * 128 + c]  = nv;
}

// ---------------------------------------------------------------------------
// Launch
// ---------------------------------------------------------------------------
extern "C" void kernel_launch(void* const* d_in, const int* in_sizes, int n_in,
                              void* d_out, int out_size, void* d_ws, size_t ws_size,
                              hipStream_t stream)
{
    (void)in_sizes; (void)n_in; (void)out_size; (void)ws_size;
    const float* x   = (const float*)d_in[0];
    const float* scl = (const float*)d_in[49];
    auto P = [&](int i) { return (const float*)d_in[i]; };
    float* out = (float*)d_out;

    // bump allocator over workspace
    char* wsp = (char*)d_ws;
    size_t off = 0;
    auto walloc = [&](size_t bytes) -> char* {
        char* p = wsp + off;
        off = (off + bytes + 255) & ~(size_t)255;
        return p;
    };

    _Float16* xh   = (_Float16*)walloc((size_t)Bn * HW * 2);
    _Float16* bufA = (_Float16*)walloc((size_t)Bn * HW * 256 * 2);
    _Float16* bufB = (_Float16*)walloc((size_t)Bn * HW * 256 * 2);
    float* logits  = (float*)walloc((size_t)Bn * HW * 4);
    float* rawd    = (float*)walloc((size_t)Bn * HW * 128 * 4);
    float* nmsbuf  = (float*)walloc((size_t)Bn * HW * 4);
    unsigned* hist = (unsigned*)walloc((size_t)Bn * 2048 * 4);
    int* ctrl      = (int*)walloc((size_t)Bn * 8 * 4);
    int* tidx      = (int*)walloc((size_t)Bn * TOPK * 4);
    float* tval    = (float*)walloc((size_t)Bn * TOPK * 4);

    struct WB { _Float16* wh; float* sc; float* bs; int coutp, kpad; };
    auto wballoc = [&](int coutp, int kpad) -> WB {
        WB w;
        w.wh = (_Float16*)walloc((size_t)coutp * kpad * 2);
        w.sc = (float*)walloc((size_t)coutp * 4);
        w.bs = (float*)walloc((size_t)coutp * 4);
        w.coutp = coutp; w.kpad = kpad;
        return w;
    };
    WB w1 = wballoc(64, 32),     w2 = wballoc(64, 576);
    WB w3 = wballoc(128, 576),   w4 = wballoc(128, 1152);
    WB wd3 = wballoc(256, 1152), wd1 = wballoc(16, 256);
    WB we3 = wballoc(256, 1152), we1 = wballoc(128, 256);

    auto prep = [&](int base, WB& w, int cin, int cout, int ks) {
        int total = w.coutp * w.kpad;
        k_prep_weights<<<(total + 255) / 256, 256, 0, stream>>>(
            P(base), P(base + 1), P(base + 2), P(base + 3), P(base + 4), P(base + 5),
            w.wh, w.sc, w.bs, cin, cout, ks, w.coutp, w.kpad);
    };
    prep(1,  w1, 1,   64,  3);
    prep(7,  w2, 64,  64,  3);
    prep(13, w3, 64,  128, 3);
    prep(19, w4, 128, 128, 3);
    prep(25, wd3, 128, 256, 3);
    prep(31, wd1, 256, 1,   1);
    prep(37, we3, 128, 256, 3);
    prep(43, we1, 256, 128, 1);

    k_x_to_h<<<(Bn * HW + 255) / 256, 256, 0, stream>>>(x, xh);

    dim3 cg(HW / 128, 1, Bn);  // 1040 x 1 x 2, 256 threads (8 wave32)
    k_conv_wmma<1,   64,  3, true,  false><<<cg, 256, 0, stream>>>(xh,   w1.wh,  w1.sc,  w1.bs,  bufA, nullptr);
    k_conv_wmma<64,  64,  3, true,  false><<<cg, 256, 0, stream>>>(bufA, w2.wh,  w2.sc,  w2.bs,  bufB, nullptr);
    k_conv_wmma<64,  128, 3, true,  false><<<cg, 256, 0, stream>>>(bufB, w3.wh,  w3.sc,  w3.bs,  bufA, nullptr);
    k_conv_wmma<128, 128, 3, true,  false><<<cg, 256, 0, stream>>>(bufA, w4.wh,  w4.sc,  w4.bs,  bufB, nullptr);
    // feats now in bufB; detection head
    k_conv_wmma<128, 256, 3, true,  false><<<cg, 256, 0, stream>>>(bufB, wd3.wh, wd3.sc, wd3.bs, bufA, nullptr);
    k_conv_wmma<256, 1,   1, false, true ><<<cg, 256, 0, stream>>>(bufA, wd1.wh, wd1.sc, wd1.bs, nullptr, logits);
    // descriptor head
    k_conv_wmma<128, 256, 3, true,  false><<<cg, 256, 0, stream>>>(bufB, we3.wh, we3.sc, we3.bs, bufA, nullptr);
    k_conv_wmma<256, 128, 1, false, true ><<<cg, 256, 0, stream>>>(bufA, we1.wh, we1.sc, we1.bs, nullptr, rawd);

    int npix = Bn * HW;
    k_score<<<(npix + 255) / 256, 256, 0, stream>>>(logits, out + O_SCORE, out + O_DPOS);
    k_nms<<<(npix + 255) / 256, 256, 0, stream>>>(out + O_SCORE, nmsbuf);

    k_fill_f32<<<(npix + 255) / 256, 256, 0, stream>>>(out + O_NMS, 0.f, npix);
    k_fill_u32<<<(Bn * 2048 + 255) / 256, 256, 0, stream>>>(hist, 0u, Bn * 2048);
    k_fill_u32<<<1, 64, 0, stream>>>((unsigned*)ctrl, 0u, Bn * 8);

    k_hist<<<(npix + 255) / 256, 256, 0, stream>>>(nmsbuf, hist);
    k_select<<<Bn, 1, 0, stream>>>(hist, ctrl);
    k_collect<<<(npix + 255) / 256, 256, 0, stream>>>(nmsbuf, ctrl, tidx, tval, out + O_NMS);

    dim3 fg((TOPK + 255) / 256, Bn);
    k_fillrest<<<fg, 256, 0, stream>>>(ctrl, tidx, tval);
    k_sparse_pos<<<fg, 256, 0, stream>>>(tidx, tval, out + O_SPOS);
    k_sparse_desc<<<dim3(TOPK, Bn), 128, 0, stream>>>(tidx, rawd, scl, out + O_SDESC);
    k_dense_desc<<<dim3(HW, Bn), 128, 0, stream>>>(rawd, scl, out + O_NDESC, out + O_DDESC);
}
